// SpatioConvLayer_9457517986085
// MI455X (gfx1250) — matrix operand order repaired and checked by reading, not compiled, not run on hardware
//
#include <hip/hip_runtime.h>

// CDNA5 / gfx1250: wave32, WMMA 16x16x32 bf16 -> f32 accumulate.
typedef __attribute__((ext_vector_type(16))) __bf16 v16bf;
typedef __attribute__((ext_vector_type(8)))  __bf16 v8bf;
typedef __attribute__((ext_vector_type(4)))  __bf16 v4bf;
typedef __attribute__((ext_vector_type(8)))  float  v8f;

#define WMMA_BF16(A, B, C) \
    __builtin_amdgcn_wmma_f32_16x16x32_bf16(false, (A), false, (B), (short)0, (C), false, false)

// Problem constants: B=32, c_in=32, T=12, N=1024, ks=3, c_out=64
#define CI   32
#define CO   64
#define TT   12
#define NN   1024
#define KS   3
#define KK   (KS * CI)   // 96 = fused contraction dim for stage 2
#define XELEMS   (32 * CI * TT * NN)   // 12,582,912
#define LKELEMS  (KS * NN * NN)        //  3,145,728

static __device__ __forceinline__ v16bf cat8(v8bf lo, v8bf hi) {
    v16bf r;
    #pragma unroll
    for (int e = 0; e < 8; ++e) { r[e] = lo[e]; r[8 + e] = hi[e]; }
    return r;
}

// 8 contiguous fp32 -> 8 bf16 (slow path, lowers to v_cvt_pk_bf16_f32)
static __device__ __forceinline__ v8bf cvt8(const float* p) {
    const float4 a = *(const float4*)(p);
    const float4 b = *(const float4*)(p + 4);
    v8bf r;
    r[0] = (__bf16)a.x; r[1] = (__bf16)a.y; r[2] = (__bf16)a.z; r[3] = (__bf16)a.w;
    r[4] = (__bf16)b.x; r[5] = (__bf16)b.y; r[6] = (__bf16)b.z; r[7] = (__bf16)b.w;
    return r;
}

// ---- Pre-pass: fp32 -> bf16 (layout preserved; hot loop then loads bf16 directly)
__global__ __launch_bounds__(256) void cvt_bf16_kernel(const float* __restrict__ src,
                                                       __bf16* __restrict__ dst, int n4) {
    const int i = blockIdx.x * blockDim.x + threadIdx.x;
    if (i < n4) {
        const float4 f = ((const float4*)src)[i];
        v4bf o;
        o[0] = (__bf16)f.x; o[1] = (__bf16)f.y; o[2] = (__bf16)f.z; o[3] = (__bf16)f.w;
        ((v4bf*)dst)[i] = o;
    }
}

// ---- Fused STGCN kernel.
// Workgroup = 2 (b,t) pairs x 64-wide n-tile. 256 threads = 8 wave32s.
// Stage 1: x_c[g][i,k,n] = sum_m x[b,i,t,m]*Lk[k,n,m]   (bf16 WMMA, K-loop over m)
// Stage 2: out[g][o,n]  = relu(theta^T @ x_c + bias + skip)
template <bool PRE>
__global__ __launch_bounds__(256) void stgcn_fused_kernel(
    const __bf16* __restrict__ xbf,   // bf16 copy of x   (PRE only)
    const __bf16* __restrict__ lkbf,  // bf16 copy of Lk  (PRE only)
    const float* __restrict__ x,      // (32, 32, 12, 1024)
    const float* __restrict__ Lk,     // (3, 1024, 1024)
    const float* __restrict__ theta,  // (32, 64, 3)
    const float* __restrict__ bias,   // 64 floats
    float* __restrict__ out)          // (32, 64, 12, 1024)
{
    __shared__ __bf16 thA[CO * KK];       // theta as A-matrix [o][kk=k*32+i], 12 KB
    __shared__ __bf16 xcB[2][64 * KK];    // x_c as B-matrix [g][n][kk], 24 KB

    const int tid  = threadIdx.x;
    const int lane = tid & 31;
    const int w    = tid >> 5;            // wave 0..7
    const int l15  = lane & 15;
    const int hi   = (lane >> 4) & 1;

    const int blk   = blockIdx.x;         // = btpair*16 + ntile
    const int ntile = blk & 15;
    const int bt0   = (blk >> 4) * 2;
    const int n0    = ntile * 64;

    // Stage theta -> LDS bf16 A-matrix layout
    for (int idx = tid; idx < CO * KK; idx += 256) {
        const int o = idx / KK, kk = idx % KK;
        const int k = kk >> 5, i = kk & 31;
        thA[idx] = (__bf16)theta[(i * CO + o) * KS + k];
    }

    // Wave mapping: g = which (b,t) pair, ni = 16-wide n-subtile
    const int g  = w >> 2;
    const int ni = w & 3;
    const int bt = bt0 + g;
    const int t  = bt % TT;
    const int b  = bt / TT;
    const int gn = n0 + ni * 16 + l15;        // B column (n) this lane supplies
    const int kbA = hi ? 8 : 0;               // A layout: K runs {0..7,16..23} / {8..15,24..31}
    const int kbB = hi ? 8 : 0;               // B layout: K runs {0..7,16..23} / {8..15,24..31}

    const size_t xoff0 = ((size_t)(b * CI + l15) * TT + t) * NN;        // i-tile 0 row
    const size_t xoff1 = ((size_t)(b * CI + 16 + l15) * TT + t) * NN;   // i-tile 1 row
    const size_t loff  = (size_t)gn * NN;

    v8f acc[KS][2];
    #pragma unroll
    for (int k = 0; k < KS; ++k) { acc[k][0] = (v8f){}; acc[k][1] = (v8f){}; }

    for (int m0 = 0; m0 < NN; m0 += 32) {
        v16bf A0, A1;
        if constexpr (PRE) {
            const __bf16* p0 = xbf + xoff0 + m0 + kbA;
            const __bf16* p1 = xbf + xoff1 + m0 + kbA;
            A0 = cat8(*(const v8bf*)p0, *(const v8bf*)(p0 + 16));
            A1 = cat8(*(const v8bf*)p1, *(const v8bf*)(p1 + 16));
        } else {
            const float* p0 = x + xoff0 + m0 + kbA;
            const float* p1 = x + xoff1 + m0 + kbA;
            A0 = cat8(cvt8(p0), cvt8(p0 + 16));
            A1 = cat8(cvt8(p1), cvt8(p1 + 16));
        }
        #pragma unroll
        for (int k = 0; k < KS; ++k) {
            v16bf Bv;
            if constexpr (PRE) {
                const __bf16* pb = lkbf + (size_t)k * NN * NN + loff + m0 + kbB;
                Bv = cat8(*(const v8bf*)pb, *(const v8bf*)(pb + 16));
            } else {
                const float* pb = Lk + (size_t)k * NN * NN + loff + m0 + kbB;
                Bv = cat8(cvt8(pb), cvt8(pb + 16));
            }
            acc[k][0] = WMMA_BF16(A0, Bv, acc[k][0]);
            acc[k][1] = WMMA_BF16(A1, Bv, acc[k][1]);
        }
    }

    // Spill x_c to LDS in stage-2 B-matrix layout: xcB[g][n_local*96 + k*32 + i]
    {
        const int nl = ni * 16 + l15;       // C/D layout: N = lane%16
        #pragma unroll
        for (int k = 0; k < KS; ++k)
            #pragma unroll
            for (int mi = 0; mi < 2; ++mi)
                #pragma unroll
                for (int j = 0; j < 8; ++j)     // C/D layout: M = vgpr + 8*(lane>=16)
                    xcB[g][nl * KK + k * 32 + mi * 16 + j + 8 * hi] = (__bf16)acc[k][mi][j];
    }
    __syncthreads();

    // ---- Stage 2: wave keeps (g, ni); B operands loaded once, reused across 4 o-tiles.
    v16bf B2[KS];
    #pragma unroll
    for (int kc = 0; kc < KS; ++kc) {
        const __bf16* pB = &xcB[g][(ni * 16 + l15) * KK + kc * 32 + kbB];
        B2[kc] = cat8(*(const v8bf*)pB, *(const v8bf*)(pB + 16));
    }

    #pragma unroll
    for (int oi = 0; oi < 4; ++oi) {
        v8f oacc = (v8f){};
        #pragma unroll
        for (int kc = 0; kc < KS; ++kc) {
            const __bf16* pA = &thA[(oi * 16 + l15) * KK + kc * 32 + kbA];
            const v16bf A2 = cat8(*(const v8bf*)pA, *(const v8bf*)(pA + 16));
            oacc = WMMA_BF16(A2, B2[kc], oacc);
        }
        // Epilogue: bias + zero-padded channel skip + ReLU (lanes contiguous in n)
        #pragma unroll
        for (int j = 0; j < 8; ++j) {
            const int o = oi * 16 + j + 8 * hi;
            float v = oacc[j] + bias[o];
            if (o < CI)
                v += x[((size_t)(b * CI + o) * TT + t) * NN + gn];
            out[((size_t)(b * CO + o) * TT + t) * NN + gn] = v > 0.0f ? v : 0.0f;
        }
    }
}

extern "C" void kernel_launch(void* const* d_in, const int* in_sizes, int n_in,
                              void* d_out, int out_size, void* d_ws, size_t ws_size,
                              hipStream_t stream) {
    const float* x     = (const float*)d_in[0];
    const float* Lk    = (const float*)d_in[1];
    const float* theta = (const float*)d_in[2];
    const float* bias  = (const float*)d_in[3];
    float* out = (float*)d_out;

    // Grid: (32*12 bt-pairs / 2) * (1024/64 n-tiles) = 192 * 16 = 3072 WGs, 8 waves each
    const int grid = (32 * TT / 2) * (NN / 64);

    const size_t need = (size_t)(XELEMS + LKELEMS) * sizeof(__bf16);   // 31.5 MB
    if (ws_size >= need) {
        __bf16* xbf  = (__bf16*)d_ws;
        __bf16* lkbf = xbf + XELEMS;
        cvt_bf16_kernel<<<(XELEMS / 4 + 255) / 256, 256, 0, stream>>>(x, xbf, XELEMS / 4);
        cvt_bf16_kernel<<<(LKELEMS / 4 + 255) / 256, 256, 0, stream>>>(Lk, lkbf, LKELEMS / 4);
        stgcn_fused_kernel<true><<<grid, 256, 0, stream>>>(xbf, lkbf, x, Lk, theta, bias, out);
    } else {
        stgcn_fused_kernel<false><<<grid, 256, 0, stream>>>(nullptr, nullptr, x, Lk, theta, bias, out);
    }
}
